// OrderQuery_35107062677745
// MI455X (gfx1250) — matrix operand order repaired and checked by reading, not compiled, not run on hardware
//
#include <hip/hip_runtime.h>

// ---------------------------------------------------------------------------
// OrderQuery: fused  [text x query^T] -> argmax_q -> segment-sum -> argsort
// MI455X / gfx1250, wave32, WMMA bf16 split-precision GEMM (3x bf16 ~= fp32)
// Segment sums use integer atomics (weights are integers) => bit-deterministic.
// ---------------------------------------------------------------------------

typedef __attribute__((ext_vector_type(16))) __bf16 v16bf;
typedef __attribute__((ext_vector_type(8)))  float  v8f;

#define B_    8
#define Q_    256
#define T_    16384
#define D_    256
#define MTILE 128          // text rows per workgroup
#define KC_   (D_ / 32)    // 8 K-chunks of 32
#define NT_   (Q_ / 16)    // 16 N-tiles of 16 queries

__device__ __forceinline__ unsigned short bf16_rne(float f) {
    unsigned int u = __float_as_uint(f);
    u += 0x7FFFu + ((u >> 16) & 1u);          // round-to-nearest-even
    return (unsigned short)(u >> 16);
}
__device__ __forceinline__ float bf16_to_f32(unsigned short h) {
    return __uint_as_float(((unsigned int)h) << 16);
}

union Frag16 {             // 16 bf16 = 8 VGPRs, loadable as two uint4
    uint4 u[2];
    v16bf v;
};

// --- zero the atomic accumulators (order + count) each call -----------------
__global__ void zero_ws(unsigned int* __restrict__ p, int n) {
    int i = blockIdx.x * 256 + threadIdx.x;
    if (i < n) p[i] = 0u;
}

// --- split fp32 query into bf16 hi + bf16 lo --------------------------------
__global__ void split_query(const float* __restrict__ q,
                            unsigned short* __restrict__ hi,
                            unsigned short* __restrict__ lo, int n) {
    int i = blockIdx.x * 256 + threadIdx.x;
    if (i < n) {
        float a = q[i];
        unsigned short h = bf16_rne(a);
        hi[i] = h;
        lo[i] = bf16_rne(a - bf16_to_f32(h));
    }
}

// --- main fused kernel: stream text once, WMMA scores, argmax, atomics ------
__global__ void __launch_bounds__(256)
score_argmax(const float* __restrict__ text,
             const unsigned short* __restrict__ qHi,
             const unsigned short* __restrict__ qLo,
             unsigned int* __restrict__ order,
             unsigned int* __restrict__ cnt) {
    extern __shared__ unsigned short smem[];          // 2 * MTILE * D_ ushorts
    unsigned short* sHi = smem;
    unsigned short* sLo = smem + MTILE * D_;

    const int tilesPerBatch = T_ / MTILE;             // 128
    const int bb  = blockIdx.x / tilesPerBatch;
    const int m0  = (blockIdx.x % tilesPerBatch) * MTILE;
    const int tid = threadIdx.x;

    // ---- stage text tile into LDS as split bf16 (coalesced float4 loads) --
    const float* tbase = text + ((size_t)bb * T_ + m0) * D_;
    const float4* t4 = (const float4*)tbase;
    #pragma unroll 4
    for (int it = 0; it < (MTILE * D_) / (256 * 4); ++it) {
        int i4 = it * 256 + tid;
        float4 f = t4[i4];
        int e = i4 * 4;
        unsigned short h0 = bf16_rne(f.x), h1 = bf16_rne(f.y);
        unsigned short h2 = bf16_rne(f.z), h3 = bf16_rne(f.w);
        sHi[e + 0] = h0; sHi[e + 1] = h1; sHi[e + 2] = h2; sHi[e + 3] = h3;
        sLo[e + 0] = bf16_rne(f.x - bf16_to_f32(h0));
        sLo[e + 1] = bf16_rne(f.y - bf16_to_f32(h1));
        sLo[e + 2] = bf16_rne(f.z - bf16_to_f32(h2));
        sLo[e + 3] = bf16_rne(f.w - bf16_to_f32(h3));
    }
    __syncthreads();

    const int lane  = tid & 31;
    const int wv    = tid >> 5;           // 8 waves, each owns 16 t-rows
    const int nlane = lane & 15;
    const int khalf = lane >> 4;          // 0 or 1
    const int rowL  = wv * 16 + nlane;    // A-layout: row = lane % 16

    // ---- preload all A-fragments (hi+lo, 8 K-chunks) into registers -------
    // 16-bit A 16x32 layout: lane holds K = khalf*8 .. +7  and  +16 .. +23
    Frag16 ahi[KC_], alo[KC_];
    #pragma unroll
    for (int kc = 0; kc < KC_; ++kc) {
        const unsigned short* pH = &sHi[rowL * D_ + kc * 32 + khalf * 8];
        const unsigned short* pL = &sLo[rowL * D_ + kc * 32 + khalf * 8];
        ahi[kc].u[0] = *(const uint4*)(pH);
        ahi[kc].u[1] = *(const uint4*)(pH + 16);
        alo[kc].u[0] = *(const uint4*)(pL);
        alo[kc].u[1] = *(const uint4*)(pL + 16);
    }

    float bestV[8];
    int   bestN[8];
    #pragma unroll
    for (int r = 0; r < 8; ++r) { bestV[r] = -3.4e38f; bestN[r] = 0; }

    // B 32x16 layout: lane holds col N = lane%16, K = khalf*8 .. +7  and  +16 .. +23
    const size_t qoff = ((size_t)bb * Q_ + nlane) * D_ + khalf * 8;

    for (int nt = 0; nt < NT_; ++nt) {
        const unsigned short* ph = qHi + qoff + (size_t)nt * 16 * D_;
        const unsigned short* pl = qLo + qoff + (size_t)nt * 16 * D_;
        v8f c0 = {};   // ah*bh chain
        v8f c1 = {};   // ah*bl + al*bh correction chains
        #pragma unroll
        for (int kc = 0; kc < KC_; ++kc) {
            Frag16 bh, bl;
            bh.u[0] = *(const uint4*)(ph + kc * 32);
            bh.u[1] = *(const uint4*)(ph + kc * 32 + 16);
            bl.u[0] = *(const uint4*)(pl + kc * 32);
            bl.u[1] = *(const uint4*)(pl + kc * 32 + 16);
            // split-precision: ah*bh + (ah*bl + al*bh)  (~fp32 accurate)
            c0 = __builtin_amdgcn_wmma_f32_16x16x32_bf16(
                    false, ahi[kc].v, false, bh.v, (short)0, c0, false, false);
            c1 = __builtin_amdgcn_wmma_f32_16x16x32_bf16(
                    false, ahi[kc].v, false, bl.v, (short)0, c1, false, false);
            c1 = __builtin_amdgcn_wmma_f32_16x16x32_bf16(
                    false, alo[kc].v, false, bh.v, (short)0, c1, false, false);
        }
        // C/D layout: lane holds N = nt*16 + lane%16; VGPR r -> M = khalf*8 + r
        int ncol = nt * 16 + nlane;
        #pragma unroll
        for (int r = 0; r < 8; ++r) {
            float v = c0[r] + c1[r];
            if (v > bestV[r]) { bestV[r] = v; bestN[r] = ncol; }  // first-max wins
        }
    }

    // ---- reduce argmax across the 16 lanes sharing each M row -------------
    #pragma unroll
    for (int r = 0; r < 8; ++r) {
        float bv = bestV[r];
        int   bn = bestN[r];
        #pragma unroll
        for (int m = 1; m < 16; m <<= 1) {
            float ov = __shfl_xor(bv, m, 32);
            int   on = __shfl_xor(bn, m, 32);
            if (ov > bv || (ov == bv && on < bn)) { bv = ov; bn = on; }
        }
        if (nlane == 0) {
            int trow = m0 + wv * 16 + khalf * 8 + r;          // global t index
            // integer weights t..1: integer atomics => exact & deterministic
            atomicAdd(&order[bb * Q_ + bn], (unsigned int)(T_ - trow));
            atomicAdd(&cnt[bb * Q_ + bn], 1u);
        }
    }
}

// --- finalize: queryorder = q - order/(count+eps); stable argsort (rank) ----
__global__ void finalize(const unsigned int* __restrict__ order,
                         const unsigned int* __restrict__ cnt,
                         float* __restrict__ out) {
    __shared__ float vals[Q_];
    const int b  = blockIdx.x;
    const int qi = threadIdx.x;
    float num = (float)order[b * Q_ + qi];
    float den = (float)cnt[b * Q_ + qi] + 0.001f;
    float v = (float)Q_ - num / den;
    vals[qi] = v;
    __syncthreads();
    int rank = 0;
    for (int j = 0; j < Q_; ++j) {
        float vj = vals[j];
        rank += (vj < v) || (vj == v && j < qi);   // stable ascending
    }
    out[b * Q_ + rank] = (float)qi;                // permutation: all slots hit
}

extern "C" void kernel_launch(void* const* d_in, const int* in_sizes, int n_in,
                              void* d_out, int out_size, void* d_ws, size_t ws_size,
                              hipStream_t stream) {
    (void)in_sizes; (void)n_in; (void)out_size; (void)ws_size;
    const float* query = (const float*)d_in[0];   // [8, 256, 256] f32
    const float* text  = (const float*)d_in[1];   // [8, 16384, 256] f32
    float* out = (float*)d_out;                   // [8, 256] (indices as float)

    unsigned int* order = (unsigned int*)d_ws;              // [B*Q]
    unsigned int* cntp  = order + B_ * Q_;                  // [B*Q]
    unsigned short* qHi = (unsigned short*)(cntp + B_ * Q_);     // [B*Q*D] bf16
    unsigned short* qLo = qHi + (size_t)B_ * Q_ * D_;            // [B*Q*D] bf16

    zero_ws<<<(2 * B_ * Q_ + 255) / 256, 256, 0, stream>>>(order, 2 * B_ * Q_);
    split_query<<<(B_ * Q_ * D_) / 256, 256, 0, stream>>>(query, qHi, qLo,
                                                          B_ * Q_ * D_);
    score_argmax<<<B_ * (T_ / MTILE), 256,
                   2 * MTILE * D_ * (int)sizeof(unsigned short), stream>>>(
        text, qHi, qLo, order, cntp);
    finalize<<<B_, 256, 0, stream>>>(order, cntp, out);
}
